// SparseMemoryLayer_55551107006693
// MI455X (gfx1250) — matrix operand order repaired and checked by reading, not compiled, not run on hardware
//
#include <hip/hip_runtime.h>
#include <hip/hip_bf16.h>
#include <math.h>

typedef __attribute__((ext_vector_type(16))) _Float16 v16h;
typedef __attribute__((ext_vector_type(8)))  float    v8f;

#define B_   32
#define IN_  256
#define H_   512
#define N_   32768
#define W_   64
#define K_   8
#define EPS_ 1e-12f
#define TPB  256
#define CHUNK 2048
#define NEG_INF_ (-3.0e38f)

__device__ __forceinline__ float sigm(float x) { return 1.0f / (1.0f + __expf(-x)); }

// Fragment index helpers (16-bit A 16x32 and B 32x16 layouts, wave32):
//   A elem e (lane L): K = 16*(L/16) + e,                        M = L%16
//   B elem e (lane L): K = 16*(L/16) + e,                        N = L%16
__device__ __forceinline__ int a_kidx(int e, int half) {
  return (half << 4) + e;
}

// ---------------------------------------------------------------------------
// K1: gates = [x|r_prev] @ W_ih^T + h_prev @ W_hh^T + b_ih + b_hh
// One wave per 16x16 tile; WMMA f32<-f16; DUAL accumulators (2 K-chunks per
// iteration) so back-to-back v_wmma are hazard-independent (no v_nop stall).
// grid (2048/16, 32/16) = (128, 2), block 32.
// ---------------------------------------------------------------------------
__global__ __launch_bounds__(32)
void k_gates(const float* __restrict__ x, const float* __restrict__ r_prev,
             const float* __restrict__ h_prev,
             const float* __restrict__ W_ih, const float* __restrict__ W_hh,
             const float* __restrict__ b_ih, const float* __restrict__ b_hh,
             float* __restrict__ gates)
{
  const int lane = threadIdx.x;
  const int half = lane >> 4, l16 = lane & 15;
  const int tn = blockIdx.x * 16, tm = blockIdx.y * 16;
  const int arow = tm + l16, bcol = tn + l16;
  v8f acc0 = {}, acc1 = {};
  // part 1: lstm_in (K = 320 = 5 x 64) * W_ih^T
  for (int kb = 0; kb < IN_ + W_; kb += 64) {
    v16h a0, b0, a1, b1;
#pragma unroll
    for (int e = 0; e < 16; ++e) {
      int ka0 = kb + a_kidx(e, half);
      int ka1 = ka0 + 32;
      float av0 = (ka0 < IN_) ? x[arow * IN_ + ka0] : r_prev[arow * W_ + (ka0 - IN_)];
      float av1 = (ka1 < IN_) ? x[arow * IN_ + ka1] : r_prev[arow * W_ + (ka1 - IN_)];
      a0[e] = (_Float16)av0;
      a1[e] = (_Float16)av1;
      int kbk = kb + (half << 4) + e;
      b0[e] = (_Float16)W_ih[bcol * (IN_ + W_) + kbk];
      b1[e] = (_Float16)W_ih[bcol * (IN_ + W_) + kbk + 32];
    }
    acc0 = __builtin_amdgcn_wmma_f32_16x16x32_f16(false, a0, false, b0, (short)0, acc0, false, false);
    acc1 = __builtin_amdgcn_wmma_f32_16x16x32_f16(false, a1, false, b1, (short)0, acc1, false, false);
  }
  // part 2: h_prev (K = 512 = 8 x 64) * W_hh^T
  for (int kb = 0; kb < H_; kb += 64) {
    v16h a0, b0, a1, b1;
#pragma unroll
    for (int e = 0; e < 16; ++e) {
      int ka0 = kb + a_kidx(e, half);
      a0[e] = (_Float16)h_prev[arow * H_ + ka0];
      a1[e] = (_Float16)h_prev[arow * H_ + ka0 + 32];
      int kbk = kb + (half << 4) + e;
      b0[e] = (_Float16)W_hh[bcol * H_ + kbk];
      b1[e] = (_Float16)W_hh[bcol * H_ + kbk + 32];
    }
    acc0 = __builtin_amdgcn_wmma_f32_16x16x32_f16(false, a0, false, b0, (short)0, acc0, false, false);
    acc1 = __builtin_amdgcn_wmma_f32_16x16x32_f16(false, a1, false, b1, (short)0, acc1, false, false);
  }
#pragma unroll
  for (int d = 0; d < 8; ++d) {
    int row = tm + (half << 3) + d, col = tn + l16;
    gates[row * (4 * H_) + col] = (acc0[d] + acc1[d]) + b_ih[col] + b_hh[col];
  }
}

// ---------------------------------------------------------------------------
// K2: LSTM pointwise -> h_curr, c_curr
// ---------------------------------------------------------------------------
__global__ void k_lstm(const float* __restrict__ gates, const float* __restrict__ c_prev,
                       float* __restrict__ h_out, float* __restrict__ c_out)
{
  int t = blockIdx.x * blockDim.x + threadIdx.x;
  if (t >= B_ * H_) return;
  int b = t >> 9, j = t & (H_ - 1);
  const float* g = gates + b * 4 * H_;
  float gi = g[j], gf = g[j + H_], gg = g[j + 2 * H_], go = g[j + 3 * H_];
  float c = sigm(gf) * c_prev[t] + sigm(gi) * tanhf(gg);
  float h = sigm(go) * tanhf(c);
  c_out[t] = c;
  h_out[t] = h;
}

// ---------------------------------------------------------------------------
// K3: params = h_curr @ W_proj^T + b_proj  ([32, 194], K=512) — tiny
// ---------------------------------------------------------------------------
__global__ void k_proj(const float* __restrict__ h, const float* __restrict__ W_proj,
                       const float* __restrict__ b_proj, float* __restrict__ params)
{
  int b = blockIdx.x, j = threadIdx.x;
  if (j >= 3 * W_ + 2) return;
  const float* hv = h + b * H_;
  const float* wrow = W_proj + j * H_;
  float s = b_proj[j];
  for (int k = 0; k < H_; ++k) s += hv[k] * wrow[k];
  params[b * (3 * W_ + 2) + j] = s;
}

// ---------------------------------------------------------------------------
// K4: cosine sim over M (268 MB streamed, float4/row pass) + per-chunk top-8
// grid (16 chunks, 32 batches), block 256.
// ---------------------------------------------------------------------------
__global__ __launch_bounds__(TPB)
void k_sim(const float* __restrict__ M, const float* __restrict__ params,
           float* __restrict__ cval, int* __restrict__ cidx)
{
  __shared__ __align__(16) float s_q[W_];
  __shared__ float s_sim[CHUNK];
  __shared__ float s_rv[TPB];
  __shared__ int   s_ri[TPB];
  const int tid = threadIdx.x, chunk = blockIdx.x, b = blockIdx.y;
  if (tid < W_) s_q[tid] = params[b * (3 * W_ + 2) + tid];
  __syncthreads();
  float qss = 0.f;
#pragma unroll
  for (int w = 0; w < W_; ++w) qss += s_q[w] * s_q[w];
  const float qn = fmaxf(sqrtf(qss), EPS_);
  const float4* q4 = (const float4*)s_q;
  for (int i = 0; i < CHUNK / TPB; ++i) {
    int nl = tid + i * TPB;
    int n = chunk * CHUNK + nl;
    const float4* row = (const float4*)(M + ((size_t)(b * N_ + n)) * W_);
    float dot = 0.f, ss = 0.f;
#pragma unroll
    for (int w4 = 0; w4 < W_ / 4; ++w4) {
      float4 v = row[w4];
      float4 q = q4[w4];
      dot += v.x * q.x + v.y * q.y + v.z * q.z + v.w * q.w;
      ss  += v.x * v.x + v.y * v.y + v.z * v.z + v.w * v.w;
    }
    s_sim[nl] = dot / (qn * fmaxf(sqrtf(ss), EPS_));
  }
  __syncthreads();
  for (int it = 0; it < K_; ++it) {
    float mv = NEG_INF_; int mi = 0;
#pragma unroll
    for (int i = 0; i < CHUNK / TPB; ++i) {
      int j = tid + i * TPB;
      float v = s_sim[j];
      if (v > mv) { mv = v; mi = j; }
    }
    s_rv[tid] = mv; s_ri[tid] = mi;
    __syncthreads();
    for (int s = TPB / 2; s > 0; s >>= 1) {
      if (tid < s && s_rv[tid + s] > s_rv[tid]) { s_rv[tid] = s_rv[tid + s]; s_ri[tid] = s_ri[tid + s]; }
      __syncthreads();
    }
    if (tid == 0) {
      int wi = s_ri[0];
      cval[(b * 16 + chunk) * K_ + it] = s_rv[0];
      cidx[(b * 16 + chunk) * K_ + it] = chunk * CHUNK + wi;
      s_sim[wi] = NEG_INF_;
    }
    __syncthreads();
  }
}

// ---------------------------------------------------------------------------
// K5: per-batch global top-8 over 128 candidates + softmax + r_curr gather
// grid 32, block 32 (one wave per batch).
// ---------------------------------------------------------------------------
__global__ __launch_bounds__(32)
void k_topk(const float* __restrict__ cval, const int* __restrict__ cidx,
            const float* __restrict__ M,
            int* __restrict__ tidx_out, float* __restrict__ tsm_out,
            float* __restrict__ r_out)
{
  __shared__ float s_v[128];
  __shared__ int   s_i[128];
  __shared__ float s_rv[32];
  __shared__ int   s_ri[32];
  __shared__ float s_tv[K_];
  __shared__ int   s_ti[K_];
  __shared__ float s_sm[K_];
  const int b = blockIdx.x, tid = threadIdx.x;
  for (int j = tid; j < 128; j += 32) { s_v[j] = cval[b * 128 + j]; s_i[j] = cidx[b * 128 + j]; }
  __syncthreads();
  for (int it = 0; it < K_; ++it) {
    float mv = NEG_INF_; int mi = 0;
    for (int j = tid; j < 128; j += 32) { if (s_v[j] > mv) { mv = s_v[j]; mi = j; } }
    s_rv[tid] = mv; s_ri[tid] = mi;
    __syncthreads();
    for (int s = 16; s > 0; s >>= 1) {
      if (tid < s && s_rv[tid + s] > s_rv[tid]) { s_rv[tid] = s_rv[tid + s]; s_ri[tid] = s_ri[tid + s]; }
      __syncthreads();
    }
    if (tid == 0) {
      int wj = s_ri[0];
      s_tv[it] = s_rv[0];
      s_ti[it] = s_i[wj];
      s_v[wj] = NEG_INF_;
    }
    __syncthreads();
  }
  if (tid == 0) {
    float mx = s_tv[0], sum = 0.f;
    for (int k = 0; k < K_; ++k) { float e = __expf(s_tv[k] - mx); s_sm[k] = e; sum += e; }
    float inv = 1.f / sum;
    for (int k = 0; k < K_; ++k) {
      s_sm[k] *= inv;
      tidx_out[b * K_ + k] = s_ti[k];
      tsm_out[b * K_ + k]  = s_sm[k];
    }
  }
  __syncthreads();
  for (int w = tid; w < W_; w += 32) {
    float acc = 0.f;
    for (int k = 0; k < K_; ++k)
      acc += s_sm[k] * M[((size_t)(b * N_ + s_ti[k])) * W_ + w];
    r_out[b * W_ + w] = acc;
  }
}

// ---------------------------------------------------------------------------
// K6: usage_curr + scattered w_r + per-chunk argmax (accessed auto-excluded:
// accessed usage=0 < usage_prev+1 >= 1). grid (16, 32), block 256.
// ---------------------------------------------------------------------------
__global__ __launch_bounds__(TPB)
void k_usage(const float* __restrict__ usage_prev,
             const int* __restrict__ tidx, const float* __restrict__ tsm,
             float* __restrict__ usage_out, float* __restrict__ wr_out,
             float* __restrict__ amv, int* __restrict__ ami)
{
  __shared__ float s_rv[TPB];
  __shared__ int   s_ri[TPB];
  const int tid = threadIdx.x, chunk = blockIdx.x, b = blockIdx.y;
  int ti[K_]; float ts[K_];
#pragma unroll
  for (int k = 0; k < K_; ++k) { ti[k] = tidx[b * K_ + k]; ts[k] = tsm[b * K_ + k]; }
  float mu = -1.f; int mn = 0;
  for (int i = 0; i < CHUNK / TPB; ++i) {
    int n = chunk * CHUNK + tid + i * TPB;
    float u = usage_prev[b * N_ + n] + 1.0f;
    float wr = 0.f; bool acc = false;
#pragma unroll
    for (int k = 0; k < K_; ++k) if (n == ti[k]) { acc = true; wr = ts[k]; }
    if (acc) u = 0.f;
    usage_out[b * N_ + n] = u;
    wr_out[b * N_ + n] = wr;
    if (u > mu) { mu = u; mn = n; }
  }
  s_rv[tid] = mu; s_ri[tid] = mn;
  __syncthreads();
  for (int s = TPB / 2; s > 0; s >>= 1) {
    if (tid < s && s_rv[tid + s] > s_rv[tid]) { s_rv[tid] = s_rv[tid + s]; s_ri[tid] = s_ri[tid + s]; }
    __syncthreads();
  }
  if (tid == 0) { amv[b * 16 + chunk] = s_rv[0]; ami[b * 16 + chunk] = s_ri[0]; }
}

// ---------------------------------------------------------------------------
// K7: per-batch LRU + write-gate scalars + precompute sigmoid(e), a vectors
// grid 32, block 64.
// ---------------------------------------------------------------------------
__global__ __launch_bounds__(64)
void k_lru(const float* __restrict__ amv, const int* __restrict__ ami,
           const float* __restrict__ params,
           int* __restrict__ lru, float* __restrict__ wwg, float* __restrict__ wwl,
           float* __restrict__ esig, float* __restrict__ avec)
{
  const int b = blockIdx.x, tid = threadIdx.x;
  if (tid == 0) {
    float mv = -1.f; int mi = 0;
    for (int c = 0; c < 16; ++c) {
      float v = amv[b * 16 + c];
      if (v > mv) { mv = v; mi = ami[b * 16 + c]; }
    }
    lru[b] = mi;
    float alpha = sigm(params[b * (3 * W_ + 2) + 3 * W_]);
    float gamma = sigm(params[b * (3 * W_ + 2) + 3 * W_ + 1]);
    wwg[b] = alpha * gamma;
    wwl[b] = alpha * (1.f - gamma);
  }
  esig[b * W_ + tid] = sigm(params[b * (3 * W_ + 2) + 2 * W_ + tid]);
  avec[b * W_ + tid] = params[b * (3 * W_ + 2) + W_ + tid];
}

// ---------------------------------------------------------------------------
// K8: M_curr = M*(1 - w_w*e) + w_w*a — 536 MB streamed, float4, 16 thr/row.
// grid 65536, block 256 (16 rows per block, fully contiguous 4KB/block).
// ---------------------------------------------------------------------------
__global__ __launch_bounds__(TPB)
void k_mupd(const float* __restrict__ M, const float* __restrict__ wr_prev,
            const int* __restrict__ lru, const float* __restrict__ wwg,
            const float* __restrict__ wwl, const float* __restrict__ esig,
            const float* __restrict__ avec, float* __restrict__ M_out)
{
  const int tid = threadIdx.x;
  const int row = blockIdx.x * (TPB / 16) + (tid >> 4);   // 0 .. B*N-1
  const int b = row >> 15, n = row & (N_ - 1);
  const int w4 = tid & 15;
  const float ww = wwg[b] * wr_prev[b * N_ + n] + ((n == lru[b]) ? wwl[b] : 0.f);
  float4 m = ((const float4*)(M + (size_t)row * W_))[w4];
  float4 e = ((const float4*)(esig + b * W_))[w4];
  float4 a = ((const float4*)(avec + b * W_))[w4];
  float4 o;
  o.x = m.x * (1.f - ww * e.x) + ww * a.x;
  o.y = m.y * (1.f - ww * e.y) + ww * a.y;
  o.z = m.z * (1.f - ww * e.z) + ww * a.z;
  o.w = m.w * (1.f - ww * e.w) + ww * a.w;
  ((float4*)(M_out + (size_t)row * W_))[w4] = o;
}

// ---------------------------------------------------------------------------
// K9: y_out = [h_curr | r_curr] @ W_final^T + b_final — WMMA tiles, dual acc.
// grid (512/16, 32/16) = (32, 2), block 32.  K = 576 = 9 x 64.
// ---------------------------------------------------------------------------
__global__ __launch_bounds__(32)
void k_yout(const float* __restrict__ h, const float* __restrict__ r,
            const float* __restrict__ W_final, const float* __restrict__ b_final,
            float* __restrict__ y)
{
  const int lane = threadIdx.x;
  const int half = lane >> 4, l16 = lane & 15;
  const int tn = blockIdx.x * 16, tm = blockIdx.y * 16;
  const int arow = tm + l16, bcol = tn + l16;
  v8f acc0 = {}, acc1 = {};
  for (int kb = 0; kb < H_ + W_; kb += 64) {
    v16h a0, b0, a1, b1;
#pragma unroll
    for (int e = 0; e < 16; ++e) {
      int ka0 = kb + a_kidx(e, half);
      int ka1 = ka0 + 32;
      float av0 = (ka0 < H_) ? h[arow * H_ + ka0] : r[arow * W_ + (ka0 - H_)];
      float av1 = (ka1 < H_) ? h[arow * H_ + ka1] : r[arow * W_ + (ka1 - H_)];
      a0[e] = (_Float16)av0;
      a1[e] = (_Float16)av1;
      int kbk = kb + (half << 4) + e;
      b0[e] = (_Float16)W_final[bcol * (H_ + W_) + kbk];
      b1[e] = (_Float16)W_final[bcol * (H_ + W_) + kbk + 32];
    }
    acc0 = __builtin_amdgcn_wmma_f32_16x16x32_f16(false, a0, false, b0, (short)0, acc0, false, false);
    acc1 = __builtin_amdgcn_wmma_f32_16x16x32_f16(false, a1, false, b1, (short)0, acc1, false, false);
  }
#pragma unroll
  for (int d = 0; d < 8; ++d) {
    int row = tm + (half << 3) + d, col = tn + l16;
    y[row * H_ + col] = (acc0[d] + acc1[d]) + b_final[col];
  }
}

// ---------------------------------------------------------------------------
extern "C" void kernel_launch(void* const* d_in, const int* in_sizes, int n_in,
                              void* d_out, int out_size, void* d_ws, size_t ws_size,
                              hipStream_t stream)
{
  const float* x       = (const float*)d_in[0];
  const float* h_prev  = (const float*)d_in[1];
  const float* c_prev  = (const float*)d_in[2];
  const float* M_prev  = (const float*)d_in[3];
  const float* wr_prev = (const float*)d_in[4];
  const float* usage_p = (const float*)d_in[5];
  const float* r_prev  = (const float*)d_in[6];
  const float* W_ih    = (const float*)d_in[7];
  const float* W_hh    = (const float*)d_in[8];
  const float* b_ih    = (const float*)d_in[9];
  const float* b_hh    = (const float*)d_in[10];
  const float* W_proj  = (const float*)d_in[11];
  const float* b_proj  = (const float*)d_in[12];
  const float* W_final = (const float*)d_in[13];
  const float* b_final = (const float*)d_in[14];

  float* out   = (float*)d_out;
  float* y_out = out;                                    // 32*512
  float* h_out = out + 16384;                            // 32*512
  float* c_out = out + 32768;                            // 32*512
  float* M_out = out + 49152;                            // 32*32768*64
  float* wr_out = M_out + (size_t)B_ * N_ * W_;          // 32*32768
  float* us_out = wr_out + (size_t)B_ * N_;              // 32*32768
  float* r_out  = us_out + (size_t)B_ * N_;              // 32*64

  float* ws     = (float*)d_ws;
  float* gates  = ws;                  // 65536
  float* params = ws + 65536;          // 6208 (pad 6272)
  float* cval   = ws + 71808;          // 32*16*8 = 4096
  int*   cidx   = (int*)(ws + 75904);  // 4096
  int*   tidx   = (int*)(ws + 80000);  // 256
  float* tsm    = ws + 80256;          // 256
  float* amv    = ws + 80512;          // 512
  int*   ami    = (int*)(ws + 81024);  // 512
  int*   lru    = (int*)(ws + 81536);  // 32
  float* wwg    = ws + 81568;          // 32
  float* wwl    = ws + 81600;          // 32
  float* esig   = ws + 81632;          // 2048 (16B aligned)
  float* avec   = ws + 83680;          // 2048 (16B aligned)

  k_gates<<<dim3(128, 2), 32, 0, stream>>>(x, r_prev, h_prev, W_ih, W_hh, b_ih, b_hh, gates);
  k_lstm <<<64, 256, 0, stream>>>(gates, c_prev, h_out, c_out);
  k_proj <<<32, 256, 0, stream>>>(h_out, W_proj, b_proj, params);
  k_sim  <<<dim3(16, 32), TPB, 0, stream>>>(M_prev, params, cval, cidx);
  k_topk <<<32, 32, 0, stream>>>(cval, cidx, M_prev, tidx, tsm, r_out);
  k_usage<<<dim3(16, 32), TPB, 0, stream>>>(usage_p, tidx, tsm, us_out, wr_out, amv, ami);
  k_lru  <<<32, 64, 0, stream>>>(amv, ami, params, lru, wwg, wwl, esig, avec);
  k_mupd <<<B_ * N_ / 16, TPB, 0, stream>>>(M_prev, wr_prev, lru, wwg, wwl, esig, avec, M_out);
  k_yout <<<dim3(32, 2), 32, 0, stream>>>(h_out, r_out, W_final, b_final, y_out);
}
